// VQVAE2_72052371358037
// MI455X (gfx1250) — matrix operand order repaired and checked by reading, not compiled, not run on hardware
//
#include <hip/hip_runtime.h>

typedef _Float16 v16h __attribute__((ext_vector_type(16)));
typedef float    v8f  __attribute__((ext_vector_type(8)));

union Frag16 { v16h v; _Float16 h[16]; uint4 q[2]; };

// ---------------------------------------------------------------------------
// f32 -> f16 convert (input image)
// ---------------------------------------------------------------------------
__global__ void cvt_f32_f16(const float* __restrict__ src, _Float16* __restrict__ dst, long n) {
    long i = (long)blockIdx.x * blockDim.x + threadIdx.x;
    if (i < n) dst[i] = (_Float16)src[i];
}

// ---------------------------------------------------------------------------
// Pack HWIO f32 weights -> f16 [tap][coutp][cin_pad32] (B-operand friendly:
// per lane a K-chunk is 32 contiguous halves).
// ---------------------------------------------------------------------------
__global__ void pack_weights(const float* __restrict__ src, _Float16* __restrict__ dst,
                             int taps, int Cin, int Cout, int Coutp, int KcPad) {
    int i = blockIdx.x * blockDim.x + threadIdx.x;
    int total = taps * Coutp * KcPad;
    if (i >= total) return;
    int c = i % KcPad;
    int n = (i / KcPad) % Coutp;
    int t = i / (KcPad * Coutp);
    float v = 0.f;
    if (c < Cin && n < Cout) v = src[((long)t * Cin + c) * Cout + n];
    dst[i] = (_Float16)v;
}

// ---------------------------------------------------------------------------
// Codebook squared norms: e2[k] = sum_d emb[d][k]^2   (emb is D x K, K=512)
// ---------------------------------------------------------------------------
__global__ void emb_norms(const float* __restrict__ emb, float* __restrict__ e2, int D, int K) {
    int k = blockIdx.x * blockDim.x + threadIdx.x;
    if (k >= K) return;
    float s = 0.f;
    for (int d = 0; d < D; ++d) { float v = emb[(long)d * K + k]; s = fmaf(v, v, s); }
    e2[k] = s;
}

// ---------------------------------------------------------------------------
// Vector quantization (forward == codebook gather of argmin distance).
// One thread per spatial vector, D=32, K=512. Codebook reads are wave-uniform
// broadcasts -> L0-resident.
// ---------------------------------------------------------------------------
__global__ void vq_kernel(const _Float16* __restrict__ in, const float* __restrict__ emb,
                          const float* __restrict__ e2, _Float16* __restrict__ out, long N) {
    long i = (long)blockIdx.x * blockDim.x + threadIdx.x;
    if (i >= N) return;
    float f[32];
#pragma unroll
    for (int d = 0; d < 32; ++d) f[d] = (float)in[i * 32 + d];
    float best = 3.4e38f;
    int bi = 0;
    for (int k = 0; k < 512; ++k) {
        float dot = 0.f;
#pragma unroll
        for (int d = 0; d < 32; ++d) dot = fmaf(f[d], emb[d * 512 + k], dot);
        float s = e2[k] - 2.f * dot;   // |f|^2 constant per row -> dropped
        if (s < best) { best = s; bi = k; }
    }
#pragma unroll
    for (int d = 0; d < 32; ++d) out[i * 32 + d] = (_Float16)emb[d * 512 + bi];
}

// ---------------------------------------------------------------------------
// Implicit-GEMM conv / transposed-conv, fully specialized per layer.
//  - one wave: 16(flattened NHW positions) x 16(out channels) tile
//  - fully unrolled taps x K-chunks of v_wmma_f32_16x16x32_f16
//  - all index math on constexpr powers of two -> shifts/masks, no div
//  - optional channel-concat input (in1 C1 | in2 C2), fused epilogue
// ACT: 0 none, 1 relu, 2 leaky(0.3), 3 sigmoid
// ---------------------------------------------------------------------------
template <int IH, int IW, int OH, int OW, int C1, int C2, int COUT,
          int KH, int KW, int S, int PAD, bool TR, int ACT, bool BN, bool RES, bool O32>
__global__ __launch_bounds__(256) void conv_wmma(
    const _Float16* __restrict__ in1, const _Float16* __restrict__ in2,
    const _Float16* __restrict__ Wp,          // [tap][COUTP][KCPAD] f16
    const float* __restrict__ bias,
    const float* __restrict__ gamma, const float* __restrict__ beta,
    const _Float16* __restrict__ residual,
    _Float16* __restrict__ out16, float* __restrict__ out32)
{
    constexpr int CINT  = C1 + C2;
    constexpr int KCPAD = (CINT + 31) & ~31;
    constexpr int COUTP = (COUT + 15) & ~15;
    constexpr int NT    = COUTP / 16;
    constexpr int P     = 8 * OH * OW;        // always a multiple of 16
    constexpr int POST  = P / 16;
    constexpr bool VEC  = ((C1 % 8) == 0) && ((C2 % 8) == 0) && ((CINT % 32) == 0);

    const int lane = threadIdx.x & 31;
    const int wv   = (int)((blockIdx.x * (long)blockDim.x + threadIdx.x) >> 5);
    if (wv >= POST * NT) return;              // wave-uniform exit
    const int nt  = wv % NT;                  // NT is 1/2/4 -> mask
    const int pt  = wv / NT;
    const int m   = lane & 15;
    const int grp = lane >> 4;

    const int p  = pt * 16 + m;               // output position (A row)
    const int b  = p / (OH * OW);             // shifts (powers of two)
    const int oy = (p / OW) % OH;
    const int ox = p % OW;
    const int n  = nt * 16 + m;               // output channel (B col / D col)

    v8f acc = {};

#pragma unroll
    for (int ky = 0; ky < KH; ++ky) {
#pragma unroll
        for (int kx = 0; kx < KW; ++kx) {
            const int t = ky * KW + kx;
            int iy, ix;
            bool vt = true;
            if constexpr (!TR) {
                iy = oy * S + ky - PAD;
                ix = ox * S + kx - PAD;
            } else {
                const int ty = oy + PAD - ky, tx = ox + PAD - kx;
                if constexpr (S > 1)
                    vt = ((ty & (S - 1)) == 0) && ((tx & (S - 1)) == 0) && ty >= 0 && tx >= 0;
                iy = ty / S; ix = tx / S;
            }
            vt = vt && iy >= 0 && iy < IH && ix >= 0 && ix < IW;
            const long sp = ((long)b * IH + iy) * IW + ix;
            const _Float16* a1 = in1 + sp * C1;
            const _Float16* a2 = (C2 > 0) ? in2 + sp * C2 : nullptr;
            const _Float16* wrow = Wp + ((long)t * COUTP + n) * KCPAD;

#pragma unroll
            for (int c0 = 0; c0 < CINT; c0 += 32) {
                // ---- A fragment: ISA wave32 16-bit A 16x32 layout ----
                // grp0: halves 0-7 -> K 0-7,  halves 8-15 -> K 16-23
                // grp1: halves 0-7 -> K 8-15, halves 8-15 -> K 24-31
                Frag16 A;
#pragma unroll
                for (int seg = 0; seg < 2; ++seg) {
                    const int cs = c0 + grp * 8 + seg * 16;   // multiple of 8
                    if constexpr (VEC) {
                        // segment never straddles the C1|C2 split (both mult of 8)
                        const _Float16* src = (cs < C1) ? (a1 + cs) : (a2 + (cs - C1));
                        if (vt) A.q[seg] = *(const uint4*)src;
                        else    { A.q[seg].x = 0u; A.q[seg].y = 0u; A.q[seg].z = 0u; A.q[seg].w = 0u; }
                    } else {
#pragma unroll
                        for (int j = 0; j < 8; ++j) {
                            const int c = cs + j;
                            _Float16 v = (_Float16)0.f;
                            if (vt && c < CINT) v = (c < C1) ? a1[c] : a2[c - C1];
                            A.h[seg * 8 + j] = v;
                        }
                    }
                }
                // ---- B fragment: lane col = lane&15, K same as A (8*grp + half, +16 for halves 8-15) ----
                Frag16 Bf;
                Bf.q[0] = *(const uint4*)(wrow + c0 + grp * 8);
                Bf.q[1] = *(const uint4*)(wrow + c0 + grp * 8 + 16);

                acc = __builtin_amdgcn_wmma_f32_16x16x32_f16(
                    false, A.v, false, Bf.v, (short)0, acc, false, false);
            }
        }
    }

    // ---- epilogue: D layout lane col = lane&15, row r -> M = r + 8*grp ----
    const bool nv = (COUT == COUTP) ? true : (n < COUT);
    const float bi = nv ? bias[n] : 0.f;
    float g = 1.f, be = 0.f;
    if constexpr (BN) { if (nv) { g = gamma[n] * rsqrtf(1.f + 1e-3f); be = beta[n]; } }
#pragma unroll
    for (int r = 0; r < 8; ++r) {
        const long po = (long)pt * 16 + r + grp * 8;
        if (!nv) continue;
        float v = acc[r] + bi;
        if constexpr (BN) v = v * g + be;
        if constexpr (RES) v += (float)residual[po * COUT + n];
        if constexpr (ACT == 1) v = fmaxf(v, 0.f);
        else if constexpr (ACT == 2) v = (v > 0.f) ? v : 0.3f * v;
        else if constexpr (ACT == 3) v = 1.f / (1.f + __expf(-v));
        if constexpr (O32) out32[po * COUT + n] = v;
        else               out16[po * COUT + n] = (_Float16)v;
    }
}

// ---------------------------------------------------------------------------
// Host-side typed launcher
// ---------------------------------------------------------------------------
template <int IH, int IW, int OH, int OW, int C1, int C2, int COUT,
          int KH, int KW, int S, int PAD, bool TR, int ACT, bool BN, bool RES, bool O32>
static void run_conv(const void* in1, const void* in2, const void* wp,
                     const float* bias, const float* gamma, const float* beta,
                     const void* res, void* out16, float* out32, hipStream_t stream) {
    constexpr int COUTP = (COUT + 15) & ~15;
    constexpr long P = 8L * OH * OW;
    constexpr long waves = (P / 16) * (COUTP / 16);
    const int blocks = (int)((waves + 7) / 8);
    conv_wmma<IH, IW, OH, OW, C1, C2, COUT, KH, KW, S, PAD, TR, ACT, BN, RES, O32>
        <<<blocks, 256, 0, stream>>>(
            (const _Float16*)in1, (const _Float16*)in2, (const _Float16*)wp,
            bias, gamma, beta, (const _Float16*)res, (_Float16*)out16, out32);
}

extern "C" void kernel_launch(void* const* d_in, const int* in_sizes, int n_in,
                              void* d_out, int out_size, void* d_ws, size_t ws_size,
                              hipStream_t stream) {
    if (n_in < 67) return;
    (void)in_sizes; (void)out_size; (void)ws_size;

    // Flattened input indices (setup_inputs dict order)
    const int IX_X = 0;
    const int IX_EB_W1 = 1, IX_EB_B1 = 2, IX_EB_W2 = 3, IX_EB_B2 = 4;
    const int IX_EBR0 = 5, IX_EBR1 = 15;            // w1,b1,g1,be1,w2,b2,g2,be2,ws,bs
    const int IX_ET_W1 = 25, IX_ET_B1 = 26, IX_ET_W2 = 27, IX_ET_B2 = 28;
    const int IX_ETR0 = 29, IX_ETR1 = 39;
    const int IX_EMB_T = 49, IX_EMB_B = 50;
    const int IX_CB_WT1 = 51, IX_CB_BT1 = 52, IX_CB_W2 = 53, IX_CB_B2 = 54;
    const int IX_D_WT1 = 55, IX_D_BT1 = 56, IX_D_WT2 = 57, IX_D_BT2 = 58;
    const int IX_D_G1 = 59, IX_D_BE1 = 60, IX_D_WT3 = 61, IX_D_BT3 = 62;
    const int IX_D_G2 = 63, IX_D_BE2 = 64, IX_D_WT4 = 65, IX_D_BT4 = 66;

    char* ws = (char*)d_ws;
    size_t off = 0;
    auto alloc = [&](size_t bytes) { size_t r = off; off += (bytes + 255) & ~(size_t)255; return r; };
    auto Fp = [&](int i) { return (const float*)d_in[i]; };

    // ---- pack all conv weights to f16 operand layout ----
    auto packW = [&](int idx, int taps, int Cin, int Cout) {
        int Coutp = (Cout + 15) & ~15;
        int KcPad = (Cin + 31) & ~31;
        size_t nelem = (size_t)taps * Coutp * KcPad;
        size_t o = alloc(nelem * sizeof(_Float16));
        pack_weights<<<(int)((nelem + 255) / 256), 256, 0, stream>>>(
            Fp(idx), (_Float16*)(ws + o), taps, Cin, Cout, Coutp, KcPad);
        return o;
    };

    size_t W_eb1 = packW(IX_EB_W1, 16, 3, 64);
    size_t W_eb2 = packW(IX_EB_W2, 16, 64, 32);
    size_t RBw1[4], RBw2[4], RBws[4];
    const int rbase[4] = { IX_EBR0, IX_EBR1, IX_ETR0, IX_ETR1 };
    for (int r = 0; r < 4; ++r) {
        RBw1[r] = packW(rbase[r] + 0, 9, 32, 32);
        RBw2[r] = packW(rbase[r] + 4, 9, 32, 32);
        RBws[r] = packW(rbase[r] + 8, 1, 32, 32);
    }
    size_t W_et1 = packW(IX_ET_W1, 16, 32, 64);
    size_t W_et2 = packW(IX_ET_W2, 9, 64, 32);
    size_t W_cb1 = packW(IX_CB_WT1, 9, 32, 64);
    size_t W_cb2 = packW(IX_CB_W2, 9, 96, 32);
    size_t W_d1  = packW(IX_D_WT1, 9, 32, 64);
    size_t W_d2  = packW(IX_D_WT2, 9, 96, 64);
    size_t W_d3  = packW(IX_D_WT3, 9, 64, 32);
    size_t W_d4  = packW(IX_D_WT4, 9, 32, 3);

    // ---- codebook norms ----
    size_t E2T = alloc(512 * sizeof(float));
    size_t E2B = alloc(512 * sizeof(float));
    emb_norms<<<2, 256, 0, stream>>>(Fp(IX_EMB_T), (float*)(ws + E2T), 32, 512);
    emb_norms<<<2, 256, 0, stream>>>(Fp(IX_EMB_B), (float*)(ws + E2B), 32, 512);

    // ---- activation buffers (f16) ----
    size_t A_X    = alloc((size_t)8 * 256 * 256 * 3  * 2);
    size_t A_E1   = alloc((size_t)8 * 128 * 128 * 64 * 2);   // reused as Z1
    size_t A_HB   = alloc((size_t)8 * 64 * 64 * 32   * 2);
    size_t A_T1   = alloc((size_t)8 * 64 * 64 * 32   * 2);   // resblock scratch
    size_t A_T2   = alloc((size_t)8 * 64 * 64 * 32   * 2);
    size_t A_HT1  = alloc((size_t)8 * 32 * 32 * 64   * 2);
    size_t A_HT   = alloc((size_t)8 * 32 * 32 * 32   * 2);
    size_t A_ETOP = alloc((size_t)8 * 32 * 32 * 32   * 2);
    size_t A_UP   = alloc((size_t)8 * 64 * 64 * 64   * 2);   // reused as XT
    size_t A_HBC  = alloc((size_t)8 * 64 * 64 * 32   * 2);
    size_t A_EBOT = alloc((size_t)8 * 64 * 64 * 32   * 2);
    size_t A_Z2   = alloc((size_t)8 * 256 * 256 * 32 * 2);
    size_t A_Z1 = A_E1;      // E1 dead after eb conv2
    size_t A_XT = A_UP;      // UP dead after cb merge conv

    // ---- network ----
    cvt_f32_f16<<<6144, 256, 0, stream>>>(Fp(IX_X), (_Float16*)(ws + A_X), 1572864L);

    // bottom encoder: 256 -> 128 -> 64
    run_conv<256,256,128,128, 3,0,64, 4,4,2,1,false, 1,false,false,false>(
        ws + A_X, nullptr, ws + W_eb1, Fp(IX_EB_B1), nullptr, nullptr, nullptr, ws + A_E1, nullptr, stream);
    run_conv<128,128,64,64, 64,0,32, 4,4,2,1,false, 0,false,false,false>(
        ws + A_E1, nullptr, ws + W_eb2, Fp(IX_EB_B2), nullptr, nullptr, nullptr, ws + A_HB, nullptr, stream);

    for (int r = 0; r < 2; ++r) {   // bottom resblocks @ 64x64
        int bb = rbase[r];
        run_conv<64,64,64,64, 32,0,32, 3,3,1,1,false, 1,true,false,false>(
            ws + A_HB, nullptr, ws + RBw1[r], Fp(bb+1), Fp(bb+2), Fp(bb+3), nullptr, ws + A_T1, nullptr, stream);
        run_conv<64,64,64,64, 32,0,32, 3,3,1,1,false, 0,true,false,false>(
            ws + A_T1, nullptr, ws + RBw2[r], Fp(bb+5), Fp(bb+6), Fp(bb+7), nullptr, ws + A_T2, nullptr, stream);
        run_conv<64,64,64,64, 32,0,32, 1,1,1,0,false, 1,false,true,false>(
            ws + A_T2, nullptr, ws + RBws[r], Fp(bb+9), nullptr, nullptr, ws + A_HB, ws + A_HB, nullptr, stream);
    }

    // top encoder: 64 -> 32
    run_conv<64,64,32,32, 32,0,64, 4,4,2,1,false, 1,false,false,false>(
        ws + A_HB, nullptr, ws + W_et1, Fp(IX_ET_B1), nullptr, nullptr, nullptr, ws + A_HT1, nullptr, stream);
    run_conv<32,32,32,32, 64,0,32, 3,3,1,1,false, 0,false,false,false>(
        ws + A_HT1, nullptr, ws + W_et2, Fp(IX_ET_B2), nullptr, nullptr, nullptr, ws + A_HT, nullptr, stream);

    for (int r = 2; r < 4; ++r) {   // top resblocks @ 32x32
        int bb = rbase[r];
        run_conv<32,32,32,32, 32,0,32, 3,3,1,1,false, 1,true,false,false>(
            ws + A_HT, nullptr, ws + RBw1[r], Fp(bb+1), Fp(bb+2), Fp(bb+3), nullptr, ws + A_T1, nullptr, stream);
        run_conv<32,32,32,32, 32,0,32, 3,3,1,1,false, 0,true,false,false>(
            ws + A_T1, nullptr, ws + RBw2[r], Fp(bb+5), Fp(bb+6), Fp(bb+7), nullptr, ws + A_T2, nullptr, stream);
        run_conv<32,32,32,32, 32,0,32, 1,1,1,0,false, 1,false,true,false>(
            ws + A_T2, nullptr, ws + RBws[r], Fp(bb+9), nullptr, nullptr, ws + A_HT, ws + A_HT, nullptr, stream);
    }

    // VQ top
    vq_kernel<<<32, 256, 0, stream>>>((const _Float16*)(ws + A_HT), Fp(IX_EMB_T),
                                      (const float*)(ws + E2T), (_Float16*)(ws + A_ETOP), 8192L);

    // conditional bottom: convT(e_top) s2 -> concat(hb|up) convT s1 -> VQ
    run_conv<32,32,64,64, 32,0,64, 3,3,2,1,true, 0,false,false,false>(
        ws + A_ETOP, nullptr, ws + W_cb1, Fp(IX_CB_BT1), nullptr, nullptr, nullptr, ws + A_UP, nullptr, stream);
    run_conv<64,64,64,64, 32,64,32, 3,3,1,1,true, 0,false,false,false>(
        ws + A_HB, ws + A_UP, ws + W_cb2, Fp(IX_CB_B2), nullptr, nullptr, nullptr, ws + A_HBC, nullptr, stream);
    vq_kernel<<<128, 256, 0, stream>>>((const _Float16*)(ws + A_HBC), Fp(IX_EMB_B),
                                       (const float*)(ws + E2B), (_Float16*)(ws + A_EBOT), 32768L);

    // decoder
    run_conv<32,32,64,64, 32,0,64, 3,3,2,1,true, 0,false,false,false>(
        ws + A_ETOP, nullptr, ws + W_d1, Fp(IX_D_BT1), nullptr, nullptr, nullptr, ws + A_XT, nullptr, stream);
    run_conv<64,64,128,128, 64,32,64, 3,3,2,1,true, 2,true,false,false>(
        ws + A_XT, ws + A_EBOT, ws + W_d2, Fp(IX_D_BT2), Fp(IX_D_G1), Fp(IX_D_BE1), nullptr, ws + A_Z1, nullptr, stream);
    run_conv<128,128,256,256, 64,0,32, 3,3,2,1,true, 2,true,false,false>(
        ws + A_Z1, nullptr, ws + W_d3, Fp(IX_D_BT3), Fp(IX_D_G2), Fp(IX_D_BE2), nullptr, ws + A_Z2, nullptr, stream);
    run_conv<256,256,256,256, 32,0,3, 3,3,1,1,true, 3,false,false,true>(
        ws + A_Z2, nullptr, ws + W_d4, Fp(IX_D_BT4), nullptr, nullptr, nullptr, nullptr, (float*)d_out, stream);
}
